// Attention_20194936226134
// MI455X (gfx1250) — matrix operand order repaired and checked by reading, not compiled, not run on hardware
//
#include <hip/hip_runtime.h>
#include <hip/hip_bf16.h>

// ---- CDNA5 wave32 WMMA types -------------------------------------------------
typedef __bf16 bf16x16 __attribute__((ext_vector_type(16)));
typedef __bf16 bf16x8  __attribute__((ext_vector_type(8)));
typedef __bf16 bf16x4  __attribute__((ext_vector_type(4)));
typedef float  f32x8   __attribute__((ext_vector_type(8)));
typedef unsigned int u32x4 __attribute__((ext_vector_type(4)));
typedef int          i32x4 __attribute__((ext_vector_type(4)));
typedef int          i32x8 __attribute__((ext_vector_type(8)));

#define WMMA_BF16(A, Bm, Cm) \
  __builtin_amdgcn_wmma_f32_16x16x32_bf16(false, (A), false, (Bm), (short)0, (Cm), false, false)

// Problem constants (match reference)
#define BB 4
#define TT 2048
#define CC 1024
#define HH 16
#define DD 64
#define MM (BB * TT)   // 8192
#define FF (3 * CC)    // 3072

// ---------------------------------------------------------------------------
// CDNA5 async global->LDS copy (GLOBAL_LOAD_ASYNC_TO_LDS_B128, ASYNCcnt).
// ---------------------------------------------------------------------------
__device__ __forceinline__ void async_ld_b128(void* lds, const void* g) {
  unsigned l = (unsigned)(unsigned long long)lds;
  asm volatile("global_load_async_to_lds_b128 %0, %1, off"
               :: "v"(l), "v"(g) : "memory");
}
__device__ __forceinline__ void async_ld32(void* lds, const void* g) {
  async_ld_b128(lds, g);
  async_ld_b128((char*)lds + 16, (const char*)g + 16);
}
__device__ __forceinline__ void wait_async0() {
#if __has_builtin(__builtin_amdgcn_s_wait_asynccnt)
  __builtin_amdgcn_s_wait_asynccnt(0);
#else
  asm volatile("s_wait_asynccnt 0x0" ::: "memory");
#endif
}

// ---------------------------------------------------------------------------
// CDNA5 Tensor Data Mover: 2-D tile DMA global->LDS (TENSORcnt).
// D# group layouts per CDNA5 ISA. bf16 elements.
// Issue once per workgroup (wave 0); completion = s_wait_tensorcnt + barrier.
// ---------------------------------------------------------------------------
__device__ __forceinline__ void tdm_load_tile_bf16(const void* gtile, unsigned lds_addr,
                                                   int tile_w, int tile_h,
                                                   int stride_elems) {
  unsigned long long ga = (unsigned long long)gtile;
  u32x4 g0;
  g0[0] = 1u;                                   // count=1, user descriptor
  g0[1] = lds_addr;                             // LDS byte address
  g0[2] = (unsigned)ga;                         // global_addr[31:0]
  g0[3] = (unsigned)((ga >> 32) & 0x01ffffffu)  // global_addr[56:32]
          | (2u << 30);                         // type=2 ("image")
  i32x8 g1;
  g1[0] = 1 << 16;                              // data_size=1 (2B), wg_mask=0
  g1[1] = (tile_w & 0xffff) << 16;              // tensor_dim0[15:0]  (bits 63:48)
  g1[2] = ((unsigned)tile_w >> 16)              // tensor_dim0[31:16]
          | ((tile_h & 0xffff) << 16);          // tensor_dim1[15:0]
  g1[3] = ((unsigned)tile_h >> 16)              // tensor_dim1[31:16]
          | ((tile_w & 0xffff) << 16);          // tile_dim0
  g1[4] = tile_h & 0xffff;                      // tile_dim1 (tile_dim2=0)
  g1[5] = stride_elems;                         // tensor_dim0_stride[31:0]
  g1[6] = 0;                                    // stride hi / dim1_stride lo
  g1[7] = 0;
  i32x4 z4 = (i32x4){};
#if __clang_major__ >= 23
  i32x8 z8 = (i32x8){};
  __builtin_amdgcn_tensor_load_to_lds(g0, g1, z4, z4, z8, 0);
#else
  __builtin_amdgcn_tensor_load_to_lds(g0, g1, z4, z4, 0);
#endif
}
__device__ __forceinline__ void wait_tensor0() {
#if __has_builtin(__builtin_amdgcn_s_wait_tensorcnt)
  __builtin_amdgcn_s_wait_tensorcnt(0);
#else
  asm volatile("s_wait_tensorcnt 0x0" ::: "memory");
#endif
}
__device__ __forceinline__ unsigned lds_off(const void* p) {
  return (unsigned)(unsigned long long)p;
}

// ---------------------------------------------------------------------------
// Kernel 0: fp32 -> bf16 conversion
// ---------------------------------------------------------------------------
__global__ __launch_bounds__(256)
void cvt_f32_to_bf16(const float* __restrict__ src, __bf16* __restrict__ dst, int n) {
  int i = (blockIdx.x * 256 + threadIdx.x) * 4;
  if (i + 3 < n) {
    float4 f = *(const float4*)(src + i);
    bf16x4 h;
    h[0] = (__bf16)f.x; h[1] = (__bf16)f.y; h[2] = (__bf16)f.z; h[3] = (__bf16)f.w;
    *(bf16x4*)(dst + i) = h;
  }
}

// 16x32 bf16 A-fragment from a row-major row pointer (ISA 7.12.2)
__device__ __forceinline__ bf16x16 make_a_frag(const __bf16* rowp, int hl) {
  bf16x8 lo = *(const bf16x8*)(rowp + hl * 16);
  bf16x8 hi = *(const bf16x8*)(rowp + hl * 16 + 8);
  bf16x16 a;
#pragma unroll
  for (int i = 0; i < 8; ++i) { a[i] = lo[i]; a[i + 8] = hi[i]; }
  return a;
}

// ---------------------------------------------------------------------------
// Kernel 1: QKV projection GEMM (Y = X * Wattn^T) + fused RoPE epilogue.
// 128x128 workgroup tile, 8 waves 4(M)x2(N), each wave 32x64 (2x4 WMMA accs).
// K-slab depth 64; TDM double-buffered LDS staging (one descriptor per operand).
// ---------------------------------------------------------------------------
__global__ __launch_bounds__(256)
void qkv_rope_kernel(const __bf16* __restrict__ Xb, const __bf16* __restrict__ Wb,
                     __bf16* __restrict__ Qo, __bf16* __restrict__ Ko,
                     __bf16* __restrict__ Vo) {
  __shared__ __bf16 Xs[2][128 * 64];   // 32 KB
  __shared__ __bf16 Ws[2][128 * 64];   // 32 KB

  const int tid  = threadIdx.x;
  const int wave = tid >> 5;
  const int lane = tid & 31;
  const int l16  = lane & 15;
  const int hl   = lane >> 4;
  const int waveM = wave >> 1;   // 0..3
  const int waveN = wave & 1;    // 0..1
  const int tileM = blockIdx.x * 128;
  const int tileN = blockIdx.y * 128;

  f32x8 acc[2][4];
#pragma unroll
  for (int i = 0; i < 2; ++i)
#pragma unroll
    for (int j = 0; j < 4; ++j) acc[i][j] = (f32x8){};

  // prologue: TDM-fill buffer 0 (issued by wave 0 only; EXEC-independent)
  if (wave == 0) {
    tdm_load_tile_bf16(&Xb[(size_t)tileM * CC], lds_off(&Xs[0][0]), 64, 128, CC);
    tdm_load_tile_bf16(&Wb[(size_t)tileN * CC], lds_off(&Ws[0][0]), 64, 128, CC);
  }

  for (int k0 = 0; k0 < CC; k0 += 64) {
    const int buf = (k0 >> 6) & 1;
    wait_tensor0();                       // free for waves 1..7 (TENSORcnt==0)
    __syncthreads();                      // slab resident for all waves
    if (k0 + 64 < CC && wave == 0) {      // overlap next slab's DMA with WMMAs
      tdm_load_tile_bf16(&Xb[(size_t)tileM * CC + k0 + 64],
                         lds_off(&Xs[buf ^ 1][0]), 64, 128, CC);
      tdm_load_tile_bf16(&Wb[(size_t)tileN * CC + k0 + 64],
                         lds_off(&Ws[buf ^ 1][0]), 64, 128, CC);
    }
#pragma unroll
    for (int kc = 0; kc < 2; ++kc) {
      bf16x16 afr[2], bfr[4];
#pragma unroll
      for (int tm = 0; tm < 2; ++tm)
        afr[tm] = make_a_frag(&Xs[buf][(waveM * 32 + tm * 16 + l16) * 64 + kc * 32], hl);
#pragma unroll
      for (int tn = 0; tn < 4; ++tn)
        bfr[tn] = *(const bf16x16*)(&Ws[buf][(waveN * 64 + tn * 16 + l16) * 64 +
                                             kc * 32 + hl * 16]);
#pragma unroll
      for (int tm = 0; tm < 2; ++tm)
#pragma unroll
        for (int tn = 0; tn < 4; ++tn)
          acc[tm][tn] = WMMA_BF16(afr[tm], bfr[tn], acc[tm][tn]);
    }
    __syncthreads();
  }

  // Epilogue with fused interleaved-pair RoPE on q,k sections.
  const float LN1E4_D = 9.210340371976184f / 64.0f;  // ln(10000)/d
#pragma unroll
  for (int tm = 0; tm < 2; ++tm) {
    const int mbase = tileM + waveM * 32 + tm * 16 + hl * 8;
#pragma unroll
    for (int tn = 0; tn < 4; ++tn) {
      const int n   = tileN + waveN * 64 + tn * 16 + l16;
      const int sec = n >> 10;          // 0=q 1=k 2=v (uniform across wave)
      const int hh  = (n & 1023) >> 6;
      const int dd  = n & 63;
      __bf16* dst = (sec == 0) ? Qo : (sec == 1) ? Ko : Vo;
      const float inv = __expf(-(float)(dd & ~1) * LN1E4_D);
#pragma unroll
      for (int r = 0; r < 8; ++r) {
        const int m = mbase + r;
        const int t = m & (TT - 1);
        const int b = m >> 11;
        float v = acc[tm][tn][r];
        float p = __shfl_xor(v, 1, 32);  // pair partner (column n^1)
        float o = v;
        if (sec < 2) {
          float ang = (float)t * inv;
          float c = __cosf(ang), s = __sinf(ang);
          o = (dd & 1) ? (p * s + v * c) : (v * c - p * s);
        }
        dst[(((size_t)b * HH + hh) * TT + t) * DD + dd] = (__bf16)o;
      }
    }
  }
}

// ---------------------------------------------------------------------------
// Kernel 2: flash attention over one (b,h). 128 q-rows per workgroup
// (16 per wave), streaming 64-key blocks; K async double-buffered (ASYNCcnt),
// V transposed into ping-pong LDS, S and PV via bf16 WMMA.
// ---------------------------------------------------------------------------
__global__ __launch_bounds__(256)
void flash_attn_kernel(const __bf16* __restrict__ Qg, const __bf16* __restrict__ Kg,
                       const __bf16* __restrict__ Vg, __bf16* __restrict__ Yo) {
  __shared__ __bf16 Ks[2][64 * 64];    // (key, d)     16 KB
  __shared__ __bf16 Vt[2][64 * 64];    // (d, key)     16 KB
  __shared__ __bf16 Ps[8][16 * 64];    // per-wave P staging  16 KB

  const int tid  = threadIdx.x;
  const int wave = tid >> 5;
  const int lane = tid & 31;
  const int l16  = lane & 15;
  const int hl   = lane >> 4;
  const int bh   = blockIdx.y;                    // b*16 + h
  const int q0   = blockIdx.x * 128;
  const size_t base = (size_t)bh * TT * DD;
  const int qrow = q0 + wave * 16;

  // Persistent Q fragments (16 rows x 64 d = 2 K-chunks), straight from global
  bf16x16 qf[2];
#pragma unroll
  for (int kc = 0; kc < 2; ++kc)
    qf[kc] = make_a_frag(Qg + base + (size_t)(qrow + l16) * DD + kc * 32, hl);

  f32x8 o[4];
#pragma unroll
  for (int i = 0; i < 4; ++i) o[i] = (f32x8){};
  float mrow[8], lrow[8];
#pragma unroll
  for (int r = 0; r < 8; ++r) { mrow[r] = -3.0e38f; lrow[r] = 0.0f; }

  const int ldj = tid >> 2;            // 0..63
  const int ldd = (tid & 3) * 16;      // 0,16,32,48

  // prologue: async K block 0, VGPR-load V block 0
  async_ld32(&Ks[0][ldj * 64 + ldd], Kg + base + (size_t)ldj * DD + ldd);
  bf16x16 vv = *(const bf16x16*)(Vg + base + (size_t)ldj * DD + ldd);

  for (int j0 = 0; j0 < TT; j0 += 64) {
    const int cur = (j0 >> 6) & 1;
    wait_async0();                       // K tile `cur` resident in LDS
    // scatter current V block, transposed, into Vt[cur]
#pragma unroll
    for (int i = 0; i < 16; ++i) Vt[cur][(ldd + i) * 64 + ldj] = vv[i];
    __syncthreads();
    if (j0 + 64 < TT) {                  // overlap next block's loads
      async_ld32(&Ks[cur ^ 1][ldj * 64 + ldd],
                 Kg + base + (size_t)(j0 + 64 + ldj) * DD + ldd);
      vv = *(const bf16x16*)(Vg + base + (size_t)(j0 + 64 + ldj) * DD + ldd);
    }

    // S = Q * K^T  (16 x 64 per wave)
    f32x8 s[4];
#pragma unroll
    for (int jt = 0; jt < 4; ++jt) s[jt] = (f32x8){};
#pragma unroll
    for (int kc = 0; kc < 2; ++kc)
#pragma unroll
      for (int jt = 0; jt < 4; ++jt) {
        bf16x16 bfr =
            *(const bf16x16*)(&Ks[cur][(jt * 16 + l16) * 64 + kc * 32 + hl * 16]);
        s[jt] = WMMA_BF16(qf[kc], bfr, s[jt]);
      }

    // Online softmax; row r+8*hl lives in the 16 lanes of this half.
    float alpha[8];
#pragma unroll
    for (int r = 0; r < 8; ++r) {
      float mx = -3.0e38f;
#pragma unroll
      for (int jt = 0; jt < 4; ++jt) mx = fmaxf(mx, s[jt][r] * 0.125f);
#pragma unroll
      for (int off = 1; off < 16; off <<= 1) mx = fmaxf(mx, __shfl_xor(mx, off, 32));
      float mn = fmaxf(mrow[r], mx);
      alpha[r] = __expf(mrow[r] - mn);
      mrow[r] = mn;
    }
    float lsum[8];
#pragma unroll
    for (int r = 0; r < 8; ++r) lsum[r] = 0.0f;
#pragma unroll
    for (int jt = 0; jt < 4; ++jt)
#pragma unroll
      for (int r = 0; r < 8; ++r) {
        float p = __expf(s[jt][r] * 0.125f - mrow[r]);
        s[jt][r] = p;
        lsum[r] += p;
      }
#pragma unroll
    for (int r = 0; r < 8; ++r) {
      float t = lsum[r];
#pragma unroll
      for (int off = 1; off < 16; off <<= 1) t += __shfl_xor(t, off, 32);
      lrow[r] = lrow[r] * alpha[r] + t;
#pragma unroll
      for (int nt = 0; nt < 4; ++nt) o[nt][r] *= alpha[r];
    }

    // Stage P (bf16) into per-wave LDS, then O += P * V
#pragma unroll
    for (int jt = 0; jt < 4; ++jt)
#pragma unroll
      for (int r = 0; r < 8; ++r)
        Ps[wave][(r + 8 * hl) * 64 + jt * 16 + l16] = (__bf16)s[jt][r];

    bf16x16 pf[2];
#pragma unroll
    for (int kc = 0; kc < 2; ++kc)
      pf[kc] = make_a_frag(&Ps[wave][l16 * 64 + kc * 32], hl);
#pragma unroll
    for (int kc = 0; kc < 2; ++kc)
#pragma unroll
      for (int nt = 0; nt < 4; ++nt) {
        bf16x16 bfr =
            *(const bf16x16*)(&Vt[cur][(nt * 16 + l16) * 64 + kc * 32 + hl * 16]);
        o[nt] = WMMA_BF16(pf[kc], bfr, o[nt]);
      }
    __syncthreads();
  }

  // Normalize and store Y in (b, t, h*64+d) bf16 layout for the out-proj GEMM
  const int b = bh >> 4, h = bh & 15;
#pragma unroll
  for (int nt = 0; nt < 4; ++nt)
#pragma unroll
    for (int r = 0; r < 8; ++r) {
      const int t = qrow + r + 8 * hl;
      const int c = h * 64 + nt * 16 + l16;
      Yo[((size_t)b * TT + t) * CC + c] = (__bf16)(o[nt][r] / lrow[r]);
    }
}

// ---------------------------------------------------------------------------
// Kernel 3: output projection (Out = Y * Wproj^T), fp32 result to d_out.
// Same TDM double-buffered structure as kernel 1.
// ---------------------------------------------------------------------------
__global__ __launch_bounds__(256)
void proj_kernel(const __bf16* __restrict__ Yb, const __bf16* __restrict__ Wb,
                 float* __restrict__ Out) {
  __shared__ __bf16 Xs[2][128 * 64];
  __shared__ __bf16 Ws[2][128 * 64];

  const int tid  = threadIdx.x;
  const int wave = tid >> 5;
  const int lane = tid & 31;
  const int l16  = lane & 15;
  const int hl   = lane >> 4;
  const int waveM = wave >> 1;
  const int waveN = wave & 1;
  const int tileM = blockIdx.x * 128;
  const int tileN = blockIdx.y * 128;

  f32x8 acc[2][4];
#pragma unroll
  for (int i = 0; i < 2; ++i)
#pragma unroll
    for (int j = 0; j < 4; ++j) acc[i][j] = (f32x8){};

  if (wave == 0) {
    tdm_load_tile_bf16(&Yb[(size_t)tileM * CC], lds_off(&Xs[0][0]), 64, 128, CC);
    tdm_load_tile_bf16(&Wb[(size_t)tileN * CC], lds_off(&Ws[0][0]), 64, 128, CC);
  }

  for (int k0 = 0; k0 < CC; k0 += 64) {
    const int buf = (k0 >> 6) & 1;
    wait_tensor0();
    __syncthreads();
    if (k0 + 64 < CC && wave == 0) {
      tdm_load_tile_bf16(&Yb[(size_t)tileM * CC + k0 + 64],
                         lds_off(&Xs[buf ^ 1][0]), 64, 128, CC);
      tdm_load_tile_bf16(&Wb[(size_t)tileN * CC + k0 + 64],
                         lds_off(&Ws[buf ^ 1][0]), 64, 128, CC);
    }
#pragma unroll
    for (int kc = 0; kc < 2; ++kc) {
      bf16x16 afr[2], bfr[4];
#pragma unroll
      for (int tm = 0; tm < 2; ++tm)
        afr[tm] = make_a_frag(&Xs[buf][(waveM * 32 + tm * 16 + l16) * 64 + kc * 32], hl);
#pragma unroll
      for (int tn = 0; tn < 4; ++tn)
        bfr[tn] = *(const bf16x16*)(&Ws[buf][(waveN * 64 + tn * 16 + l16) * 64 +
                                             kc * 32 + hl * 16]);
#pragma unroll
      for (int tm = 0; tm < 2; ++tm)
#pragma unroll
        for (int tn = 0; tn < 4; ++tn)
          acc[tm][tn] = WMMA_BF16(afr[tm], bfr[tn], acc[tm][tn]);
    }
    __syncthreads();
  }

#pragma unroll
  for (int tm = 0; tm < 2; ++tm) {
    const int mbase = tileM + waveM * 32 + tm * 16 + hl * 8;
#pragma unroll
    for (int tn = 0; tn < 4; ++tn) {
      const int n = tileN + waveN * 64 + tn * 16 + l16;
#pragma unroll
      for (int r = 0; r < 8; ++r)
        Out[(size_t)(mbase + r) * CC + n] = acc[tm][tn][r];
    }
  }
}

// ---------------------------------------------------------------------------
extern "C" void kernel_launch(void* const* d_in, const int* in_sizes, int n_in,
                              void* d_out, int out_size, void* d_ws, size_t ws_size,
                              hipStream_t stream) {
  const float* x  = (const float*)d_in[0];   // (4, 2048, 1024)
  const float* wa = (const float*)d_in[1];   // (3072, 1024)
  const float* wp = (const float*)d_in[2];   // (1024, 1024)
  float* out = (float*)d_out;                // (4, 2048, 1024) fp32

  char* ws = (char*)d_ws;
  const size_t MB = 1u << 20;
  __bf16* xb  = (__bf16*)(ws + 0 * MB);      // 16 MB (reused as Y below)
  __bf16* wab = (__bf16*)(ws + 16 * MB);     //  6 MB
  __bf16* wpb = (__bf16*)(ws + 22 * MB);     //  2 MB
  __bf16* q   = (__bf16*)(ws + 24 * MB);     // 16 MB
  __bf16* k   = (__bf16*)(ws + 40 * MB);     // 16 MB
  __bf16* v   = (__bf16*)(ws + 56 * MB);     // 16 MB
  __bf16* y   = xb;                          // reuse: x dead after QKV GEMM

  const int nx = MM * CC, nwa = FF * CC, nwp = CC * CC;
  cvt_f32_to_bf16<<<nx  / 1024, 256, 0, stream>>>(x,  xb,  nx);
  cvt_f32_to_bf16<<<nwa / 1024, 256, 0, stream>>>(wa, wab, nwa);
  cvt_f32_to_bf16<<<nwp / 1024, 256, 0, stream>>>(wp, wpb, nwp);

  qkv_rope_kernel<<<dim3(MM / 128, FF / 128), 256, 0, stream>>>(xb, wab, q, k, v);
  flash_attn_kernel<<<dim3(TT / 128, BB * HH), 256, 0, stream>>>(q, k, v, y);
  proj_kernel<<<dim3(MM / 128, CC / 128), 256, 0, stream>>>(y, wpb, out);
}
